// APTModel_86053964743136
// MI455X (gfx1250) — matrix operand order repaired and checked by reading, not compiled, not run on hardware
//
#include <hip/hip_runtime.h>

#define B_   4
#define S_   512
#define E_   512
#define H_   8
#define DH_  64
#define HID_ 128

typedef __attribute__((ext_vector_type(16))) __bf16 v16bf;
typedef __attribute__((ext_vector_type(8)))  __bf16 v8bf;
typedef __attribute__((ext_vector_type(8)))  float  v8f;

#define WMMA_BF16(a, b, c) \
  __builtin_amdgcn_wmma_f32_16x16x32_bf16(false, (a), false, (b), (short)0, (c), false, false)

// ---------------------------------------------------------------- helpers

__device__ __forceinline__ __bf16 f2bf(float f) {
  unsigned u = __builtin_bit_cast(unsigned, f);
  u += 0x7FFFu + ((u >> 16) & 1u);              // round to nearest even
  unsigned short h = (unsigned short)(u >> 16);
  return __builtin_bit_cast(__bf16, h);
}

// A fragment: 16x32 bf16, A row-major [M, lda]. lane&15 = M row,
// lane>=16 -> K base 8; elems 0..7 -> K kb..kb+7, elems 8..15 -> K kb+16..kb+23.
__device__ __forceinline__ v16bf load_a_frag(const __bf16* A, int lda, int row,
                                             int k0, int lane) {
  int kb = ((lane >> 4) & 1) * 8;
  const v8bf* p = reinterpret_cast<const v8bf*>(A + (size_t)row * lda + (k0 + kb));
  v8bf lo = p[0];
  v8bf hi = p[2];
  v16bf a;
#pragma unroll
  for (int i = 0; i < 8; ++i) { a[i] = lo[i]; a[i + 8] = hi[i]; }
  return a;
}

// B fragment: 32x16 bf16 with B[k][n] = Wt[n][k], Wt row-major [N, ldw].
// lane&15 = N col, lane>=16 -> K base 8; elems 0..7 -> K kb..kb+7, elems 8..15 -> K kb+16..kb+23.
__device__ __forceinline__ v16bf load_bT_frag(const __bf16* Wt, int ldw, int col,
                                              int k0, int lane) {
  int kb = ((lane >> 4) & 1) * 8;
  const v8bf* p = reinterpret_cast<const v8bf*>(Wt + (size_t)col * ldw + (k0 + kb));
  v8bf lo = p[0];
  v8bf hi = p[2];
  v16bf b;
#pragma unroll
  for (int i = 0; i < 8; ++i) { b[i] = lo[i]; b[i + 8] = hi[i]; }
  return b;
}

__device__ __forceinline__ float blk_reduce_max(float v, float* red, int tid, int nt) {
  red[tid] = v; __syncthreads();
  for (int s = nt >> 1; s > 0; s >>= 1) {
    if (tid < s) red[tid] = fmaxf(red[tid], red[tid + s]);
    __syncthreads();
  }
  float r = red[0]; __syncthreads();
  return r;
}

__device__ __forceinline__ float blk_reduce_sum(float v, float* red, int tid, int nt) {
  red[tid] = v; __syncthreads();
  for (int s = nt >> 1; s > 0; s >>= 1) {
    if (tid < s) red[tid] += red[tid + s];
    __syncthreads();
  }
  float r = red[0]; __syncthreads();
  return r;
}

// ---------------------------------------------------------------- kernels

__global__ void f32_to_bf16_kernel(const float* __restrict__ in,
                                   __bf16* __restrict__ out, int n) {
  int i = blockIdx.x * blockDim.x + threadIdx.x;
  if (i < n) out[i] = f2bf(in[i]);
}

// C[m,n] = (x[m,:] . W[n,:] + bias[n]) * scale, stored bf16 as
//  mode 0: [B,H,S,DH]  (q, k)    mode 1: [B,H,DH,S]  (v, transposed)
__global__ void qkv_gemm_kernel(const __bf16* __restrict__ xbf,
                                const __bf16* __restrict__ wbf,
                                const float* __restrict__ bias,
                                __bf16* __restrict__ out,
                                float scale, int transposed) {
  int lane = threadIdx.x;
  int m0 = blockIdx.x * 16;
  int n0 = (blockIdx.y * 4 + threadIdx.y) * 16;
  int arow = m0 + (lane & 15);
  int bcol = n0 + (lane & 15);
  __builtin_prefetch(xbf + (size_t)arow * E_, 0, 0);
  v8f acc = {};
#pragma unroll
  for (int k0 = 0; k0 < E_; k0 += 32) {
    v16bf a = load_a_frag(xbf, E_, arow, k0, lane);
    v16bf b = load_bT_frag(wbf, E_, bcol, k0, lane);
    acc = WMMA_BF16(a, b, acc);
  }
  int n = n0 + (lane & 15);
  int h = n / DH_, d = n % DH_;
  float bv = bias[n];
#pragma unroll
  for (int r = 0; r < 8; ++r) {
    int m = m0 + r + ((lane >> 4) & 1) * 8;
    int bb = m / S_, s = m % S_;
    float v = (acc[r] + bv) * scale;
    size_t idx = transposed
        ? ((size_t)(bb * H_ + h) * DH_ + d) * S_ + s
        : ((size_t)(bb * H_ + h) * S_ + s) * DH_ + d;
    out[idx] = f2bf(v);
  }
}

// scores[bh,q,n] = q[bh,q,:] . k[bh,n,:]   (q already scaled), + nan_to_num/clip
__global__ void scores_kernel(const __bf16* __restrict__ qbf,
                              const __bf16* __restrict__ kbf,
                              float* __restrict__ scores) {
  int lane = threadIdx.x;
  int bh = blockIdx.z;
  int m0 = blockIdx.x * 16;
  int n0 = (blockIdx.y * 4 + threadIdx.y) * 16;
  const __bf16* Q = qbf + (size_t)bh * S_ * DH_;
  const __bf16* K = kbf + (size_t)bh * S_ * DH_;
  int arow = m0 + (lane & 15);
  int bcol = n0 + (lane & 15);
  v8f acc = {};
#pragma unroll
  for (int k0 = 0; k0 < DH_; k0 += 32) {
    v16bf a = load_a_frag(Q, DH_, arow, k0, lane);
    v16bf b = load_bT_frag(K, DH_, bcol, k0, lane);
    acc = WMMA_BF16(a, b, acc);
  }
  int n = n0 + (lane & 15);
#pragma unroll
  for (int r = 0; r < 8; ++r) {
    int m = m0 + r + ((lane >> 4) & 1) * 8;
    float s = acc[r];
    if (s != s) s = 0.0f;                        // nan -> 0
    else if (s > 3.0e38f) s = 10.0f;             // +inf -> 10
    else if (s < -3.0e38f) s = -10.0f;           // -inf -> -10
    s = fminf(fmaxf(s, -15.0f), 15.0f);
    scores[((size_t)bh * S_ + m) * S_ + n] = s;
  }
}

__global__ void mean_kernel(const float* __restrict__ scores,
                            float* __restrict__ mean) {
  size_t i = (size_t)blockIdx.x * blockDim.x + threadIdx.x;
  size_t total = (size_t)B_ * S_ * S_;
  if (i >= total) return;
  size_t b = i / ((size_t)S_ * S_);
  size_t rem = i % ((size_t)S_ * S_);
  float s = 0.0f;
#pragma unroll
  for (int h = 0; h < H_; ++h)
    s += scores[((size_t)(b * H_ + h)) * S_ * S_ + rem];
  mean[i] = s * (1.0f / H_);
}

__global__ void zero_stats_kernel(float* stats, float* coefs) {
  int t = threadIdx.x;
  if (t < 16) stats[t] = 0.0f;
  if (t < 8) coefs[t] = 0.0f;
}

// one block per (b, i) row of mean_attn; computes t0 row and per-batch sums
__global__ void transform_kernel(const float* __restrict__ mean,
                                 const float* __restrict__ w1,
                                 const float* __restrict__ b1,
                                 const float* __restrict__ w2,
                                 const float* __restrict__ b2p,
                                 float* __restrict__ t0,
                                 float* __restrict__ stats) {
  __shared__ float red[256];
  int tid = threadIdx.x;
  int bi = blockIdx.x;                // b*S + i
  int b = bi / S_;
  const float* mrow = mean + (size_t)bi * S_;
  float m0v = mrow[tid];
  float m1v = mrow[tid + 256];

  // p = softmax(clip(mean,-10,10))
  float cm0 = fminf(fmaxf(m0v, -10.0f), 10.0f);
  float cm1 = fminf(fmaxf(m1v, -10.0f), 10.0f);
  float mx = blk_reduce_max(fmaxf(cm0, cm1), red, tid, 256);
  float e0 = __expf(cm0 - mx), e1 = __expf(cm1 - mx);
  float Z = blk_reduce_sum(e0 + e1, red, tid, 256);
  float p0 = e0 / Z, p1 = e1 / Z;
  float H0 = -p0 * __logf(p0 + 1e-6f);
  float H1 = -p1 * __logf(p1 + 1e-6f);

  // Fm = softmax(3*H)
  float fmx = blk_reduce_max(fmaxf(3.0f * H0, 3.0f * H1), red, tid, 256);
  float f0 = __expf(3.0f * H0 - fmx), f1 = __expf(3.0f * H1 - fmx);
  float FZ = blk_reduce_sum(f0 + f1, red, tid, 256);
  f0 /= FZ; f1 /= FZ;

  // 1x1 conv path: auto = sum_h clamp(scaled*w1+b1, 0, 5) * w2 + b2
  float sc0 = fminf(fmaxf(fminf(fmaxf(m0v, -8.0f), 8.0f) * 0.05f, -10.0f), 10.0f);
  float sc1 = fminf(fmaxf(fminf(fmaxf(m1v, -8.0f), 8.0f) * 0.05f, -10.0f), 10.0f);
  float a0 = 0.0f, a1 = 0.0f;
  for (int hh = 0; hh < HID_; ++hh) {
    float w = w1[hh], bb = b1[hh], w2v = w2[hh];
    a0 += fminf(fmaxf(sc0 * w + bb, 0.0f), 5.0f) * w2v;
    a1 += fminf(fmaxf(sc1 * w + bb, 0.0f), 5.0f) * w2v;
  }
  float b2v = b2p[0];
  a0 = fminf(fmaxf(a0 + b2v, -5.0f), 5.0f);
  a1 = fminf(fmaxf(a1 + b2v, -5.0f), 5.0f);
  float ty0 = fminf(fmaxf(1.0f + 2.5f * a0, 0.5f), 1.5f);
  float ty1 = fminf(fmaxf(1.0f + 2.5f * a1, 0.5f), 1.5f);
  float t00 = (1.0f / (1.0f + __expf(-ty0))) * f0;
  float t01 = (1.0f / (1.0f + __expf(-ty1))) * f1;

  t0[(size_t)bi * S_ + tid] = t00;
  t0[(size_t)bi * S_ + tid + 256] = t01;

  float st  = blk_reduce_sum(t00 + t01, red, tid, 256);
  float st2 = blk_reduce_sum(t00 * t00 + t01 * t01, red, tid, 256);
  float smm = blk_reduce_sum(m0v + m1v, red, tid, 256);
  float sm2 = blk_reduce_sum(m0v * m0v + m1v * m1v, red, tid, 256);
  if (tid == 0) {
    atomicAdd(&stats[b * 4 + 0], st);
    atomicAdd(&stats[b * 4 + 1], st2);
    atomicAdd(&stats[b * 4 + 2], smm);
    atomicAdd(&stats[b * 4 + 3], sm2);
  }
}

// fold energy + std matching into t = c0[b] + c1[b] * t0
__global__ void finalize_kernel(const float* __restrict__ stats,
                                float* __restrict__ coefs) {
  int b = threadIdx.x;
  if (b >= B_) return;
  float N = (float)S_ * (float)S_;
  float st = stats[b * 4 + 0], st2 = stats[b * 4 + 1];
  float sm = stats[b * 4 + 2], sm2 = stats[b * 4 + 3];
  float mean0 = st / N, Et2 = st2 / N, om = sm / N, Em2 = sm2 / N;
  float e_o = sqrtf(fmaxf(sm2, 0.0f)) + 1e-4f;
  float e_t = sqrtf(fmaxf(st2, 0.0f)) + 1e-4f;
  float gamma = fminf(fmaxf(e_o / e_t, 0.8f), 1.2f);
  float t1mean = gamma * mean0;
  float tvar = gamma * gamma * (Et2 - mean0 * mean0);
  float t_std = sqrtf(fmaxf(tvar, 0.01f));
  float o_std = sqrtf(fmaxf(Em2 - om * om, 0.01f));
  float gdyn = fminf(fmaxf(o_std / t_std, 0.8f), 1.2f);
  coefs[2 * b] = om - gdyn * t1mean;
  coefs[2 * b + 1] = gdyn * gamma;
}

// fused: blend transform into scores, row softmax in LDS, then attn @ V (WMMA)
__global__ void softmax_av_kernel(const float* __restrict__ scores,
                                  const float* __restrict__ t0,
                                  const float* __restrict__ coefs,
                                  const __bf16* __restrict__ vt,
                                  __bf16* __restrict__ attn_out) {
  constexpr int LDSTR = 520;                    // bf16 elems; keeps 16B alignment
  __shared__ __bf16 attn_s[16 * LDSTR];
  __shared__ float red[128];
  int tid = threadIdx.y * 32 + threadIdx.x;
  int bh = blockIdx.y;
  int b = bh / H_, h = bh % H_;
  int q0 = blockIdx.x * 16;
  float c0 = coefs[2 * b], c1 = coefs[2 * b + 1];

  for (int r = 0; r < 16; ++r) {
    int q = q0 + r;
    const float* srow = scores + ((size_t)bh * S_ + q) * S_;
    const float* trow = t0 + ((size_t)b * S_ + q) * S_;
    float v[4];
    float mx = -1e30f;
#pragma unroll
    for (int jj = 0; jj < 4; ++jj) {
      int j = tid + jj * 128;
      float tt = c0 + c1 * trow[j];
      float s = fminf(fmaxf(srow[j] + 0.1f * tt, -15.0f), 15.0f);
      v[jj] = s;
      mx = fmaxf(mx, s);
    }
    mx = blk_reduce_max(mx, red, tid, 128);
    float sm = 0.0f;
#pragma unroll
    for (int jj = 0; jj < 4; ++jj) { v[jj] = __expf(v[jj] - mx); sm += v[jj]; }
    sm = blk_reduce_sum(sm, red, tid, 128);
    float inv = 1.0f / sm;
#pragma unroll
    for (int jj = 0; jj < 4; ++jj)
      attn_s[r * LDSTR + tid + jj * 128] = f2bf(v[jj] * inv);
  }
  __syncthreads();

  // attn[16,512] @ V[512,64] -> [16,64]; wave threadIdx.y owns 16 cols of DH
  int lane = threadIdx.x;
  int n0 = threadIdx.y * 16;
  int arow = lane & 15;
  int bcol = n0 + (lane & 15);
  const __bf16* V = vt + (size_t)bh * DH_ * S_;  // [DH, S] row-major (V^T)
  v8f acc = {};
#pragma unroll
  for (int k0 = 0; k0 < S_; k0 += 32) {
    v16bf a = load_a_frag(attn_s, LDSTR, arow, k0, lane);
    v16bf bfr = load_bT_frag(V, S_, bcol, k0, lane);
    acc = WMMA_BF16(a, bfr, acc);
  }
#pragma unroll
  for (int r = 0; r < 8; ++r) {
    int m = r + ((lane >> 4) & 1) * 8;
    int s = q0 + m;
    int e = h * DH_ + n0 + (lane & 15);
    attn_out[((size_t)b * S_ + s) * E_ + e] = f2bf(acc[r]);
  }
}

// out[m,n] = attn_out[m,:] . wo[n,:] + bo[n]  (fp32 output)
__global__ void proj_gemm_kernel(const __bf16* __restrict__ abf,
                                 const __bf16* __restrict__ wbf,
                                 const float* __restrict__ bias,
                                 float* __restrict__ out) {
  int lane = threadIdx.x;
  int m0 = blockIdx.x * 16;
  int n0 = (blockIdx.y * 4 + threadIdx.y) * 16;
  int arow = m0 + (lane & 15);
  int bcol = n0 + (lane & 15);
  __builtin_prefetch(abf + (size_t)arow * E_, 0, 0);
  v8f acc = {};
#pragma unroll
  for (int k0 = 0; k0 < E_; k0 += 32) {
    v16bf a = load_a_frag(abf, E_, arow, k0, lane);
    v16bf b = load_bT_frag(wbf, E_, bcol, k0, lane);
    acc = WMMA_BF16(a, b, acc);
  }
  int n = n0 + (lane & 15);
  float bv = bias[n];
#pragma unroll
  for (int r = 0; r < 8; ++r) {
    int m = m0 + r + ((lane >> 4) & 1) * 8;
    out[(size_t)m * E_ + n] = acc[r] + bv;
  }
}

// ---------------------------------------------------------------- launch

extern "C" void kernel_launch(void* const* d_in, const int* in_sizes, int n_in,
                              void* d_out, int out_size, void* d_ws, size_t ws_size,
                              hipStream_t stream) {
  (void)in_sizes; (void)n_in; (void)out_size; (void)ws_size;
  const float* x  = (const float*)d_in[0];
  const float* wq = (const float*)d_in[1];
  const float* bq = (const float*)d_in[2];
  const float* wk = (const float*)d_in[3];
  const float* bk = (const float*)d_in[4];
  const float* wv = (const float*)d_in[5];
  const float* bv = (const float*)d_in[6];
  const float* wo = (const float*)d_in[7];
  const float* bo = (const float*)d_in[8];
  const float* w1 = (const float*)d_in[9];
  const float* b1 = (const float*)d_in[10];
  const float* w2 = (const float*)d_in[11];
  const float* b2 = (const float*)d_in[12];
  float* out = (float*)d_out;

  char* ws = (char*)d_ws;
  size_t off = 0;
  auto alloc = [&](size_t bytes) -> void* {
    void* p = ws + off;
    off += (bytes + 255) & ~(size_t)255;
    return p;
  };
  const size_t NX = (size_t)B_ * S_ * E_;        // 1,048,576
  const size_t NW = (size_t)E_ * E_;             // 262,144
  const size_t NS = (size_t)B_ * H_ * S_ * S_;   // 8,388,608
  const size_t NM = (size_t)B_ * S_ * S_;        // 1,048,576

  __bf16* xbf  = (__bf16*)alloc(NX * 2);
  __bf16* wqb  = (__bf16*)alloc(NW * 2);
  __bf16* wkb  = (__bf16*)alloc(NW * 2);
  __bf16* wvb  = (__bf16*)alloc(NW * 2);
  __bf16* wob  = (__bf16*)alloc(NW * 2);
  __bf16* qbf  = (__bf16*)alloc(NX * 2);         // [B,H,S,DH]
  __bf16* kbf  = (__bf16*)alloc(NX * 2);         // [B,H,S,DH]
  __bf16* vtb  = (__bf16*)alloc(NX * 2);         // [B,H,DH,S]
  float*  scr  = (float*)alloc(NS * 4);          // scores fp32
  float*  mea  = (float*)alloc(NM * 4);          // mean over heads
  float*  t0b  = (float*)alloc(NM * 4);          // t0
  float*  stats = (float*)alloc(16 * 4);
  float*  coefs = (float*)alloc(8 * 4);
  __bf16* aob  = (__bf16*)alloc(NX * 2);         // attn output [B,S,E]
  // total workspace ~54 MB

  // 1) fp32 -> bf16 conversions
  f32_to_bf16_kernel<<<(int)((NX + 255) / 256), 256, 0, stream>>>(x, xbf, (int)NX);
  f32_to_bf16_kernel<<<(int)((NW + 255) / 256), 256, 0, stream>>>(wq, wqb, (int)NW);
  f32_to_bf16_kernel<<<(int)((NW + 255) / 256), 256, 0, stream>>>(wk, wkb, (int)NW);
  f32_to_bf16_kernel<<<(int)((NW + 255) / 256), 256, 0, stream>>>(wv, wvb, (int)NW);
  f32_to_bf16_kernel<<<(int)((NW + 255) / 256), 256, 0, stream>>>(wo, wob, (int)NW);

  // 2) QKV projections (q pre-scaled by DH^-0.5 = 0.125)
  dim3 gg((B_ * S_) / 16, E_ / 64), bb(32, 4);
  qkv_gemm_kernel<<<gg, bb, 0, stream>>>(xbf, wqb, bq, qbf, 0.125f, 0);
  qkv_gemm_kernel<<<gg, bb, 0, stream>>>(xbf, wkb, bk, kbf, 1.0f, 0);
  qkv_gemm_kernel<<<gg, bb, 0, stream>>>(xbf, wvb, bv, vtb, 1.0f, 1);

  // 3) scores = q @ k^T (+ nan_to_num/clip epilogue)
  scores_kernel<<<dim3(S_ / 16, S_ / 64, B_ * H_), bb, 0, stream>>>(qbf, kbf, scr);

  // 4) mean over heads
  mean_kernel<<<(int)((NM + 255) / 256), 256, 0, stream>>>(scr, mea);

  // 5) autopoietic transform -> t0 + per-batch sums -> affine coefs
  zero_stats_kernel<<<1, 32, 0, stream>>>(stats, coefs);
  transform_kernel<<<B_ * S_, 256, 0, stream>>>(mea, w1, b1, w2, b2, t0b, stats);
  finalize_kernel<<<1, 32, 0, stream>>>(stats, coefs);

  // 6) fused blend + softmax + attn@V
  softmax_av_kernel<<<dim3(S_ / 16, B_ * H_), bb, 0, stream>>>(scr, t0b, coefs, vtb, aob);

  // 7) output projection (fp32 out)
  proj_gemm_kernel<<<gg, bb, 0, stream>>>(aob, wob, bo, out);
}
